// CachedMultiHeadAttention_31001073942538
// MI455X (gfx1250) — matrix operand order repaired and checked by reading, not compiled, not run on hardware
//
#include <hip/hip_runtime.h>

// ---------------- problem constants ----------------
#define BATCH 4
#define S_LEN 2048
#define HEADS 16
#define DK 64
#define DM 1024

// ---------------- clang ext-vector types ----------------
typedef __attribute__((ext_vector_type(16))) __bf16      v16bf;
typedef __attribute__((ext_vector_type(8)))  float       v8f;
typedef __attribute__((ext_vector_type(4)))  unsigned    v4u;
typedef __attribute__((ext_vector_type(4)))  float       v4f;

union FragBF {
    v4u   q[2];
    v16bf v;
};

// f32 -> bf16 (RNE fptrunc; lowers to native v_cvt on gfx1250)
__device__ __forceinline__ unsigned short f2bf(float f) {
    union { __bf16 b; unsigned short u; } c;
    c.b = (__bf16)f;
    return c.u;
}

// ---------------- weight conversion ----------------
__global__ __launch_bounds__(256) void cvt_f32_bf16(const float* __restrict__ in,
                                                    unsigned short* __restrict__ out,
                                                    int n) {
    int base = blockIdx.x * 1024;
#pragma unroll
    for (int j = 0; j < 4; ++j) {
        int id = base + j * 256 + threadIdx.x;
        if (id < n) out[id] = f2bf(in[id]);
    }
}

// ---------------- generic WMMA GEMM: D = A * B^T (+epilogue) ----------------
// A: [M, Kd] row-major (f32 if AF32 else bf16), per-z stride aZ (elements)
// B: [N, Kd] row-major bf16, per-z stride bZ (elements)
// Block tile 128x128x32, 256 threads = 8 waves (2x4), wave = 64x32 (4x2 WMMA tiles).
// Double-buffered LDS; bf16 tiles staged with global_load_async_to_lds_b128
// (ASYNCcnt), f32 A tiles loaded early and converted after the WMMA phase.
// MODE 0: Q/K proj  -> bf16 head-split  [B,H,S,DK]
// MODE 1: V proj    -> bf16 head-split transposed [B,H,DK,S]
// MODE 2: scores    -> f32 * (1/8), causal mask, out [z,S,S]
// MODE 3: attn @ V  -> bf16 merged-head ctx [B,S,DM]  (z = b*H+h, N=DK)
// MODE 4: final     -> f32 + bias, out [M, DM]
template <int MODE, bool AF32>
__global__ __launch_bounds__(256) void gemm_wmma(const void* __restrict__ Aptr,
                                                 const unsigned short* __restrict__ Bptr,
                                                 const float* __restrict__ bias,
                                                 void* __restrict__ Optr,
                                                 int M, int N, int Kd,
                                                 long long aZ, long long bZ) {
    __shared__ unsigned short As[2][128][40];   // +8 pad: 80B row stride, 16B chunks
    __shared__ unsigned short Bs[2][128][40];

    const int tid = threadIdx.x;
    const int z  = blockIdx.z;
    const int m0 = blockIdx.y * 128;
    const int n0 = blockIdx.x * 128;

    // fully-masked score tiles: no compute, write -1e9 (block-uniform branch)
    if (MODE == 2) {
        if (n0 > m0 + 127) {
            float* o = (float*)Optr + (size_t)z * S_LEN * S_LEN;
            const int c  = tid & 127;
            const int r0 = tid >> 7;
#pragma unroll
            for (int rr = 0; rr < 64; ++rr) {
                int r = r0 + rr * 2;
                o[(size_t)(m0 + r) * S_LEN + (n0 + c)] = -1e9f;
            }
            return;
        }
    }

    const float*          Af = (const float*)Aptr          + (size_t)z * (AF32 ? aZ : 0);
    const unsigned short* Ah = (const unsigned short*)Aptr + (size_t)z * (AF32 ? 0 : aZ);
    const unsigned short* Bz = Bptr + (size_t)z * bZ;

    const int lane = tid & 31;
    const int w    = tid >> 5;
    const int wm   = w >> 2;       // 0..1  -> 64 rows each
    const int wn   = w & 3;        // 0..3  -> 32 cols each
    const int lrow = lane & 15;    // M (A) / N (B) within 16x16 tile
    const int lk   = lane >> 4;    // K-half selector

    v8f acc[4][2];
    const v8f zro = {0.f, 0.f, 0.f, 0.f, 0.f, 0.f, 0.f, 0.f};
#pragma unroll
    for (int i = 0; i < 4; ++i)
#pragma unroll
        for (int j = 0; j < 2; ++j) acc[i][j] = zro;

    const int arow = tid >> 1;      // 0..127: tile row staged by this thread
    const int half = tid & 1;
    const int col  = half * 16;     // 16 bf16 (32B) per thread per matrix

    // --- staging helpers ---
    auto issueB = [&](int bb, int kk) {   // async copy 32B: global -> LDS
        if (n0 + arow < N) {
            const unsigned short* bp = Bz + (size_t)(n0 + arow) * Kd + (kk + col);
            unsigned           ldsa = (unsigned)(size_t)&Bs[bb][arow][col];
            unsigned long long ga   = (unsigned long long)(size_t)bp;
            asm volatile("global_load_async_to_lds_b128 %0, %1, off\n\t"
                         "global_load_async_to_lds_b128 %0, %1, off offset:16"
                         :: "v"(ldsa), "v"(ga) : "memory");
        }
    };
    auto issueA = [&](int bb, int kk) {   // bf16 A: async copy
        const unsigned short* ap = Ah + (size_t)(m0 + arow) * Kd + (kk + col);
        unsigned           ldsa = (unsigned)(size_t)&As[bb][arow][col];
        unsigned long long ga   = (unsigned long long)(size_t)ap;
        asm volatile("global_load_async_to_lds_b128 %0, %1, off\n\t"
                     "global_load_async_to_lds_b128 %0, %1, off offset:16"
                     :: "v"(ldsa), "v"(ga) : "memory");
    };
    auto loadA = [&](int kk, v4f* f) {    // f32 A: issue global loads (no wait yet)
        const float* ap = Af + (size_t)(m0 + arow) * Kd + (kk + col);
        f[0] = *(const v4f*)(ap + 0);
        f[1] = *(const v4f*)(ap + 4);
        f[2] = *(const v4f*)(ap + 8);
        f[3] = *(const v4f*)(ap + 12);
        __builtin_prefetch(ap + 32, 0, 1);
    };
    auto storeA = [&](int bb, const v4f* f) {   // convert + ds_store
        unsigned short* d = &As[bb][arow][col];
#pragma unroll
        for (int e = 0; e < 4; ++e) d[e +  0] = f2bf(f[0][e]);
#pragma unroll
        for (int e = 0; e < 4; ++e) d[e +  4] = f2bf(f[1][e]);
#pragma unroll
        for (int e = 0; e < 4; ++e) d[e +  8] = f2bf(f[2][e]);
#pragma unroll
        for (int e = 0; e < 4; ++e) d[e + 12] = f2bf(f[3][e]);
    };

    // --- prologue: zero OOB B rows once (both buffers), stage tile 0 ---
    if (n0 + arow >= N) {
        const v4u z0 = {0u, 0u, 0u, 0u};
        *(v4u*)&Bs[0][arow][col + 0] = z0;
        *(v4u*)&Bs[0][arow][col + 8] = z0;
        *(v4u*)&Bs[1][arow][col + 0] = z0;
        *(v4u*)&Bs[1][arow][col + 8] = z0;
    }
    issueB(0, 0);
    {
        v4f p[4];
        if (AF32) { loadA(0, p); storeA(0, p); }
        else      issueA(0, 0);
    }

    int cur = 0;
    for (int k0 = 0; k0 < Kd; k0 += 32) {
        const bool hasNext = (k0 + 32) < Kd;
        const int  nb = cur ^ 1;
        v4f fN[4];
        if (hasNext) {                       // overlap next tile's copy with WMMA
            issueB(nb, k0 + 32);
            if (AF32) loadA(k0 + 32, fN);
            else      issueA(nb, k0 + 32);
        }
        // release current tile's async loads, keep next tile's in flight
        if (hasNext) {
            if (AF32) asm volatile("s_wait_asynccnt 0x2" ::: "memory");
            else      asm volatile("s_wait_asynccnt 0x4" ::: "memory");
        } else {
            asm volatile("s_wait_asynccnt 0x0" ::: "memory");
        }
        __syncthreads();                     // all waves' current tile visible

        // ---- fragments per ISA 16-bit layouts, then 8 WMMAs ----
        FragBF a[4], b[2];
#pragma unroll
        for (int i = 0; i < 4; ++i) {
            const int r = wm * 64 + i * 16 + lrow;
            a[i].q[0] = *(const v4u*)&As[cur][r][lk * 8];        // K = lk*8 .. +8
            a[i].q[1] = *(const v4u*)&As[cur][r][16 + lk * 8];   // K = 16+lk*8 .. +8
        }
#pragma unroll
        for (int j = 0; j < 2; ++j) {
            const int r = wn * 32 + j * 16 + lrow;
            b[j].q[0] = *(const v4u*)&Bs[cur][r][lk * 8];        // K = lk*8 .. +8
            b[j].q[1] = *(const v4u*)&Bs[cur][r][16 + lk * 8];   // K = 16+lk*8 .. +8
        }
#pragma unroll
        for (int i = 0; i < 4; ++i)
#pragma unroll
            for (int j = 0; j < 2; ++j)
                acc[i][j] = __builtin_amdgcn_wmma_f32_16x16x32_bf16(
                    false, a[i].v, false, b[j].v, (short)0, acc[i][j], false, false);

        if (hasNext && AF32) storeA(nb, fN); // f32 path: convert after compute
        __syncthreads();                     // protect buffer reuse
        cur = nb;
    }

    // ---- epilogue: C/D layout -> element (m = r + 8*(lane>=16), n = lane&15) ----
    const int nIn   = lane & 15;
    const int mHalf = lane >> 4;
#pragma unroll
    for (int i = 0; i < 4; ++i) {
#pragma unroll
        for (int j = 0; j < 2; ++j) {
#pragma unroll
            for (int r = 0; r < 8; ++r) {
                const int mG = m0 + wm * 64 + i * 16 + r + 8 * mHalf;
                const int nG = n0 + wn * 32 + j * 16 + nIn;
                float val = acc[i][j][r];
                if (MODE == 0) {            // Q/K: bf16 [B,H,S,DK]
                    val += bias[nG];
                    const int bb = mG >> 11, s = mG & (S_LEN - 1);
                    const int h = nG >> 6, d = nG & (DK - 1);
                    ((unsigned short*)Optr)[((((size_t)bb * HEADS + h) * S_LEN) + s) * DK + d] = f2bf(val);
                } else if (MODE == 1) {     // V: bf16 [B,H,DK,S] (transposed)
                    val += bias[nG];
                    const int bb = mG >> 11, s = mG & (S_LEN - 1);
                    const int h = nG >> 6, d = nG & (DK - 1);
                    ((unsigned short*)Optr)[((((size_t)bb * HEADS + h) * DK) + d) * S_LEN + s] = f2bf(val);
                } else if (MODE == 2) {     // scores: scale + causal mask
                    val *= 0.125f;          // 1/sqrt(DK)
                    if (nG > mG) val = -1e9f;
                    ((float*)Optr)[(size_t)z * S_LEN * S_LEN + (size_t)mG * S_LEN + nG] = val;
                } else if (MODE == 3) {     // ctx: bf16 merged heads [B,S,DM]
                    if (nG < N) {
                        const int bb = z >> 4, h = z & (HEADS - 1);
                        ((unsigned short*)Optr)[((size_t)bb * S_LEN + mG) * DM + h * DK + nG] = f2bf(val);
                    }
                } else {                    // final: f32 + bias
                    ((float*)Optr)[(size_t)mG * DM + nG] = val + bias[nG];
                }
            }
        }
    }
}

// ---------------- row softmax over 2048 elements, 256 threads/row ----------------
__global__ __launch_bounds__(256) void softmax_rows(float* __restrict__ attn) {
    __shared__ float red[8];
    float* p = attn + (size_t)blockIdx.x * S_LEN;
    const int t = threadIdx.x;
    float v[8];
    float mx = -3.4e38f;
#pragma unroll
    for (int i = 0; i < 8; ++i) {
        v[i] = p[t + i * 256];
        mx = fmaxf(mx, v[i]);
    }
#pragma unroll
    for (int off = 16; off > 0; off >>= 1) mx = fmaxf(mx, __shfl_xor(mx, off, 32));
    if ((t & 31) == 0) red[t >> 5] = mx;
    __syncthreads();
    float m2 = red[0];
#pragma unroll
    for (int i = 1; i < 8; ++i) m2 = fmaxf(m2, red[i]);
    __syncthreads();
    float s = 0.f;
#pragma unroll
    for (int i = 0; i < 8; ++i) {
        v[i] = __expf(v[i] - m2);
        s += v[i];
    }
#pragma unroll
    for (int off = 16; off > 0; off >>= 1) s += __shfl_xor(s, off, 32);
    if ((t & 31) == 0) red[t >> 5] = s;
    __syncthreads();
    float tot = 0.f;
#pragma unroll
    for (int i = 0; i < 8; ++i) tot += red[i];
    const float inv = 1.0f / tot;
#pragma unroll
    for (int i = 0; i < 8; ++i) p[t + i * 256] = v[i] * inv;
}

// ---------------- launcher ----------------
extern "C" void kernel_launch(void* const* d_in, const int* in_sizes, int n_in,
                              void* d_out, int out_size, void* d_ws, size_t ws_size,
                              hipStream_t stream) {
    (void)in_sizes; (void)n_in; (void)out_size; (void)ws_size;

    const float* query = (const float*)d_in[0];
    const float* key   = (const float*)d_in[1];
    const float* value = (const float*)d_in[2];
    // d_in[3] = mask (causal, known analytically -> unused)
    const float* wq = (const float*)d_in[4];
    const float* bq = (const float*)d_in[5];
    const float* wk = (const float*)d_in[6];
    const float* bk = (const float*)d_in[7];
    const float* wv = (const float*)d_in[8];
    const float* bv = (const float*)d_in[9];
    const float* wo = (const float*)d_in[10];
    const float* bo = (const float*)d_in[11];

    float* out   = (float*)d_out;                               // [B,S,DM]
    float* attnF = out + (size_t)BATCH * S_LEN * DM;            // [B,H,S,S]

    const size_t W_EL = (size_t)DM * DM;                        // 1M
    const size_t H_EL = (size_t)BATCH * HEADS * S_LEN * DK;     // 8.39M
    unsigned short* wq16  = (unsigned short*)d_ws;
    unsigned short* wk16  = wq16 + W_EL;
    unsigned short* wv16  = wk16 + W_EL;
    unsigned short* wo16  = wv16 + W_EL;
    unsigned short* Qh    = wo16 + W_EL;                        // [B,H,S,DK] bf16
    unsigned short* Kh    = Qh + H_EL;                          // [B,H,S,DK] bf16
    unsigned short* Vt    = Kh + H_EL;                          // [B,H,DK,S] bf16
    unsigned short* ctx16 = Vt + H_EL;                          // [B,S,DM]   bf16

    // 1) weights -> bf16
    cvt_f32_bf16<<<(int)(W_EL / 1024), 256, 0, stream>>>(wq, wq16, (int)W_EL);
    cvt_f32_bf16<<<(int)(W_EL / 1024), 256, 0, stream>>>(wk, wk16, (int)W_EL);
    cvt_f32_bf16<<<(int)(W_EL / 1024), 256, 0, stream>>>(wv, wv16, (int)W_EL);
    cvt_f32_bf16<<<(int)(W_EL / 1024), 256, 0, stream>>>(wo, wo16, (int)W_EL);

    const dim3 blk(256);
    const int  Mrows = BATCH * S_LEN;                           // 8192
    const dim3 gProj(DM / 128, Mrows / 128, 1);                 // (8,64,1)

    // 2) projections
    gemm_wmma<0, true><<<gProj, blk, 0, stream>>>(query, wq16, bq, Qh, Mrows, DM, DM, 0, 0);
    gemm_wmma<0, true><<<gProj, blk, 0, stream>>>(key,   wk16, bk, Kh, Mrows, DM, DM, 0, 0);
    gemm_wmma<1, true><<<gProj, blk, 0, stream>>>(value, wv16, bv, Vt, Mrows, DM, DM, 0, 0);

    // 3) scores = (Q K^T)/8 with causal mask, per (b,h)
    const dim3 gSc(S_LEN / 128, S_LEN / 128, BATCH * HEADS);    // (16,16,64)
    gemm_wmma<2, false><<<gSc, blk, 0, stream>>>(Qh, Kh, nullptr, attnF,
                                                 S_LEN, S_LEN, DK,
                                                 (long long)S_LEN * DK, (long long)S_LEN * DK);

    // 4) softmax rows (attn written in-place in d_out)
    softmax_rows<<<BATCH * HEADS * S_LEN, 256, 0, stream>>>(attnF);

    // 5) ctx = attn @ V, per (b,h); A = f32 attn converted on the fly
    const dim3 gAV(1, S_LEN / 128, BATCH * HEADS);              // (1,16,64)
    gemm_wmma<3, true><<<gAV, blk, 0, stream>>>(attnF, Vt, nullptr, ctx16,
                                                S_LEN, DK, S_LEN,
                                                (long long)S_LEN * S_LEN, (long long)DK * S_LEN);

    // 6) output = ctx @ wo^T + bo
    gemm_wmma<4, false><<<gProj, blk, 0, stream>>>(ctx16, wo16, bo, out, Mrows, DM, DM, 0, 0);
}
